// Seq2SeqDecoder_21638045237355
// MI455X (gfx1250) — matrix operand order repaired and checked by reading, not compiled, not run on hardware
//
#include <hip/hip_runtime.h>
#include <cstdint>
#include <cstddef>

// ---------------------------------------------------------------------------
// Problem dims (compile-time constants from the reference)
// ---------------------------------------------------------------------------
static constexpr int Bn = 256;   // batch
static constexpr int Tn = 32;    // target length
static constexpr int Sn = 128;   // source length
static constexpr int In = 256;   // input dim
static constexpr int Hn = 1024;  // hidden dim
static constexpr int On = 256;   // output dim
static constexpr int KC = In + Hn;  // concat K for gates GEMM = 1280 (mult of 32)

typedef __bf16 bf16_t;
typedef __attribute__((ext_vector_type(16))) __bf16 v16bf;
typedef __attribute__((ext_vector_type(8)))  __bf16 v8bf;
typedef __attribute__((ext_vector_type(8)))  float  v8f;

union V16U { v16bf v; v8bf h[2]; };

// ---------------------------------------------------------------------------
// WMMA operand loaders, contiguous-k 16-bit layouts (not the ISA table order) (CDNA5 ISA 7.12.2)
// ---------------------------------------------------------------------------
// A (16x32, MxK): lanes 0-15 -> rows M=0..15, VGPR0..7 = K[hi*16 .. hi*16+15]
//                 contiguous (hi = lane>>4), matching load_b_tile
__device__ __forceinline__ v16bf load_a_tile(const bf16_t* __restrict__ Ablk,
                                             int lda, int kb, int lane) {
  const int m  = lane & 15;
  const int hi = lane >> 4;
  const bf16_t* p = Ablk + (size_t)m * lda + kb + hi * 16;
  V16U u;
  u.h[0] = *(const v8bf*)(p);        // 16B -> global_load_b128
  u.h[1] = *(const v8bf*)(p + 8);    // 16B -> global_load_b128
  return u.v;
}

// B (32x16, KxN): WMMA-B column n == weight row n (we compute A * W^T with
// row-major W[N,K]).  Lane l: n = l&15, holds K[kb + hi*16 .. +15] contiguous.
__device__ __forceinline__ v16bf load_b_tile(const bf16_t* __restrict__ Wblk,
                                             int ldw, int kb, int lane) {
  const int n  = lane & 15;
  const int hi = lane >> 4;
  const bf16_t* p = Wblk + (size_t)n * ldw + kb + hi * 16;
  return *(const v16bf*)(p);         // 32B -> 2x global_load_b128
}

// ---------------------------------------------------------------------------
// Generic bf16 WMMA GEMM:  D[M,N] = [A1 | A2] * W^T + bias
//   A1: [M,K1] lda1, A2: [M,K2] lda2 (K2 may be 0), W: [N, K1+K2] row-major.
// Block = 8 waves (32x8) arranged 2(M) x 4(N): block tile 32 x 256.
// Each wave: 16x64 tile, 4 f32 accumulators, v_wmma_f32_16x16x32_bf16.
// ---------------------------------------------------------------------------
__global__ __launch_bounds__(256) void
wmma_gemm_kernel(const bf16_t* __restrict__ A1, int lda1, int K1,
                 const bf16_t* __restrict__ A2, int lda2, int K2,
                 const bf16_t* __restrict__ W,  int ldw,
                 const float*  __restrict__ bias,
                 float* __restrict__ D, int ldd)
{
  const int lane = threadIdx.x;          // 0..31
  const int wave = threadIdx.y;          // 0..7
  const int wm = wave >> 2;              // 0..1
  const int wn = wave & 3;               // 0..3
  const int m0 = blockIdx.y * 32 + wm * 16;
  const int n0 = blockIdx.x * 256 + wn * 64;

  v8f acc0 = {}, acc1 = {}, acc2 = {}, acc3 = {};

  const bf16_t* wb0 = W + (size_t)(n0 +  0) * ldw;
  const bf16_t* wb1 = W + (size_t)(n0 + 16) * ldw;
  const bf16_t* wb2 = W + (size_t)(n0 + 32) * ldw;
  const bf16_t* wb3 = W + (size_t)(n0 + 48) * ldw;

  const bf16_t* a1 = A1 + (size_t)m0 * lda1;
#pragma unroll 2
  for (int kb = 0; kb < K1; kb += 32) {
    v16bf a  = load_a_tile(a1, lda1, kb, lane);
    v16bf b0 = load_b_tile(wb0, ldw, kb, lane);
    v16bf b1 = load_b_tile(wb1, ldw, kb, lane);
    v16bf b2 = load_b_tile(wb2, ldw, kb, lane);
    v16bf b3 = load_b_tile(wb3, ldw, kb, lane);
    acc0 = __builtin_amdgcn_wmma_f32_16x16x32_bf16(false, a, false, b0, (short)0, acc0, false, false);
    acc1 = __builtin_amdgcn_wmma_f32_16x16x32_bf16(false, a, false, b1, (short)0, acc1, false, false);
    acc2 = __builtin_amdgcn_wmma_f32_16x16x32_bf16(false, a, false, b2, (short)0, acc2, false, false);
    acc3 = __builtin_amdgcn_wmma_f32_16x16x32_bf16(false, a, false, b3, (short)0, acc3, false, false);
  }

  const bf16_t* a2 = A2 + (size_t)m0 * lda2;
#pragma unroll 2
  for (int kb = 0; kb < K2; kb += 32) {
    v16bf a  = load_a_tile(a2, lda2, kb, lane);
    v16bf b0 = load_b_tile(wb0, ldw, K1 + kb, lane);
    v16bf b1 = load_b_tile(wb1, ldw, K1 + kb, lane);
    v16bf b2 = load_b_tile(wb2, ldw, K1 + kb, lane);
    v16bf b3 = load_b_tile(wb3, ldw, K1 + kb, lane);
    acc0 = __builtin_amdgcn_wmma_f32_16x16x32_bf16(false, a, false, b0, (short)0, acc0, false, false);
    acc1 = __builtin_amdgcn_wmma_f32_16x16x32_bf16(false, a, false, b1, (short)0, acc1, false, false);
    acc2 = __builtin_amdgcn_wmma_f32_16x16x32_bf16(false, a, false, b2, (short)0, acc2, false, false);
    acc3 = __builtin_amdgcn_wmma_f32_16x16x32_bf16(false, a, false, b3, (short)0, acc3, false, false);
  }

  // C/D layout: VGPR i, lanes 0-15 -> (M=i, N=lane); lanes 16-31 -> (M=i+8)
  const int hi = lane >> 4;
  const int nc = lane & 15;
  const float bs0 = bias[n0 +  0 + nc];
  const float bs1 = bias[n0 + 16 + nc];
  const float bs2 = bias[n0 + 32 + nc];
  const float bs3 = bias[n0 + 48 + nc];
#pragma unroll
  for (int i = 0; i < 8; ++i) {
    float* drow = D + (size_t)(m0 + i + 8 * hi) * ldd;
    drow[n0 +  0 + nc] = acc0[i] + bs0;
    drow[n0 + 16 + nc] = acc1[i] + bs1;
    drow[n0 + 32 + nc] = acc2[i] + bs2;
    drow[n0 + 48 + nc] = acc3[i] + bs3;
  }
}

// ---------------------------------------------------------------------------
// Prep: Wcat[n][0:256]=W_ih[n], Wcat[n][256:1280]=W_hh[n] (bf16); bsum=b_ih+b_hh
// ---------------------------------------------------------------------------
__global__ void prep_wcat_kernel(const float* __restrict__ W_ih,
                                 const float* __restrict__ W_hh,
                                 const float* __restrict__ b_ih,
                                 const float* __restrict__ b_hh,
                                 bf16_t* __restrict__ Wcat,
                                 float* __restrict__ bsum)
{
  const int idx = blockIdx.x * blockDim.x + threadIdx.x;
  if (idx < 4 * Hn) bsum[idx] = b_ih[idx] + b_hh[idx];
  if (idx >= 4 * Hn * KC) return;
  const int n = idx / KC;
  const int k = idx % KC;
  const float v = (k < In) ? W_ih[(size_t)n * In + k]
                           : W_hh[(size_t)n * Hn + (k - In)];
  Wcat[idx] = (bf16_t)v;
}

__global__ void cvt_f32_to_bf16_kernel(const float* __restrict__ src,
                                       bf16_t* __restrict__ dst, int n)
{
  const int idx = blockIdx.x * blockDim.x + threadIdx.x;
  if (idx < n) dst[idx] = (bf16_t)src[idx];
}

// shifted_target [B,T,I] -> Xbf [T,B,I] bf16 with t==0 zeroed (reference feeds
// zeros at t=0)
__global__ void prep_x_kernel(const float* __restrict__ st,
                              bf16_t* __restrict__ Xbf)
{
  const int idx = blockIdx.x * blockDim.x + threadIdx.x;
  if (idx >= Tn * Bn * In) return;
  const int t = idx / (Bn * In);
  const int r = idx % (Bn * In);
  const int b = r / In;
  const int k = r % In;
  const float v = (t == 0) ? 0.0f : st[((size_t)b * Tn + t) * In + k];
  Xbf[idx] = (bf16_t)v;
}

// ---------------------------------------------------------------------------
// LSTM cell elementwise: gates [B,4H] -> c,h (f32), h_bf, and z[:,H:2H]=h (bf16)
// ---------------------------------------------------------------------------
__device__ __forceinline__ float sigmoidf_(float x) {
  return 1.0f / (1.0f + __expf(-x));
}

__global__ __launch_bounds__(256) void
lstm_cell_kernel(const float* __restrict__ gates,
                 float* __restrict__ c, float* __restrict__ h,
                 bf16_t* __restrict__ h_bf, bf16_t* __restrict__ z_bf)
{
  const int idx = blockIdx.x * blockDim.x + threadIdx.x;   // over B*H
  const int b = idx / Hn;
  const int j = idx % Hn;
  const float* g = gates + (size_t)b * 4 * Hn;
  const float gi = sigmoidf_(g[0 * Hn + j]);
  const float gf = sigmoidf_(g[1 * Hn + j]);
  const float gg = tanhf(g[2 * Hn + j]);
  const float go = sigmoidf_(g[3 * Hn + j]);
  const float cn = gf * c[idx] + gi * gg;
  c[idx] = cn;
  const float hn = go * tanhf(cn);
  h[idx] = hn;
  const bf16_t hb = (bf16_t)hn;
  h_bf[idx] = hb;
  z_bf[(size_t)b * (2 * Hn) + Hn + j] = hb;   // second half of FC input
}

// ---------------------------------------------------------------------------
// Attention: per-batch block.  scores = enc[b] @ h[b]; softmax; probs -> out;
// ctx -> z[:,0:H] (bf16).  enc is L2-resident (134 MB < 192 MB L2).
// ---------------------------------------------------------------------------
__global__ __launch_bounds__(256) void
attn_kernel(const float* __restrict__ h, const float* __restrict__ enc,
            float* __restrict__ probs, bf16_t* __restrict__ z_bf, int t)
{
  const int b   = blockIdx.x;
  const int tid = threadIdx.x;
  __shared__ float sh_h[Hn];
  __shared__ float sc[Sn];
  __shared__ float red[Sn];

  const float* hb = h + (size_t)b * Hn;
  for (int j = tid; j < Hn; j += 256) sh_h[j] = hb[j];
  __syncthreads();

  const float* eb = enc + (size_t)b * Sn * Hn;
  if (tid < Sn) {
    const float4* e4 = (const float4*)(eb + (size_t)tid * Hn);
    const float4* h4 = (const float4*)sh_h;
    float acc = 0.0f;
#pragma unroll 4
    for (int j = 0; j < Hn / 4; ++j) {
      const float4 ev = e4[j];
      const float4 hv = h4[j];
      acc = fmaf(ev.x, hv.x, acc);
      acc = fmaf(ev.y, hv.y, acc);
      acc = fmaf(ev.z, hv.z, acc);
      acc = fmaf(ev.w, hv.w, acc);
    }
    sc[tid] = acc;
    red[tid] = acc;
  }
  __syncthreads();
  // max-reduce
  for (int off = Sn / 2; off > 0; off >>= 1) {
    if (tid < off) red[tid] = fmaxf(red[tid], red[tid + off]);
    __syncthreads();
  }
  const float mx = red[0];
  __syncthreads();
  if (tid < Sn) {
    const float e = __expf(sc[tid] - mx);
    sc[tid]  = e;
    red[tid] = e;
  }
  __syncthreads();
  // sum-reduce
  for (int off = Sn / 2; off > 0; off >>= 1) {
    if (tid < off) red[tid] += red[tid + off];
    __syncthreads();
  }
  const float inv = 1.0f / red[0];
  __syncthreads();
  if (tid < Sn) {
    const float p = sc[tid] * inv;
    sc[tid] = p;
    probs[((size_t)b * Tn + t) * Sn + tid] = p;
  }
  __syncthreads();
  // ctx[j] = sum_s p[s] * enc[b][s][j]  -> z[:,0:H] bf16
  for (int j = tid; j < Hn; j += 256) {
    float acc = 0.0f;
#pragma unroll 4
    for (int s = 0; s < Sn; ++s)
      acc = fmaf(sc[s], eb[(size_t)s * Hn + j], acc);
    z_bf[(size_t)b * (2 * Hn) + j] = (bf16_t)acc;
  }
}

// ---------------------------------------------------------------------------
// Host-side orchestration
// ---------------------------------------------------------------------------
extern "C" void kernel_launch(void* const* d_in, const int* in_sizes, int n_in,
                              void* d_out, int out_size, void* d_ws, size_t ws_size,
                              hipStream_t stream) {
  (void)in_sizes; (void)n_in; (void)out_size; (void)ws_size;
  const float* st    = (const float*)d_in[0];   // [B,T,I]
  const float* enc   = (const float*)d_in[1];   // [B,S,H]
  const float* W_ih  = (const float*)d_in[2];   // [4H,I]
  const float* W_hh  = (const float*)d_in[3];   // [4H,H]
  const float* b_ih  = (const float*)d_in[4];   // [4H]
  const float* b_hh  = (const float*)d_in[5];   // [4H]
  const float* W_fc  = (const float*)d_in[6];   // [O,2H]
  const float* b_fc  = (const float*)d_in[7];   // [O]

  float* out   = (float*)d_out;                         // [B,T,O] ...
  float* probs = (float*)d_out + (size_t)Bn * Tn * On;  // ... then [B,T,S]

  // Workspace carve-out (~24 MB)
  char* p = (char*)d_ws;
  auto alloc = [&](size_t bytes) {
    char* r = p;
    p += (bytes + 255) & ~(size_t)255;
    return r;
  };
  bf16_t* Wcat = (bf16_t*)alloc((size_t)4 * Hn * KC * sizeof(bf16_t));
  bf16_t* Wfc  = (bf16_t*)alloc((size_t)On * 2 * Hn * sizeof(bf16_t));
  float*  bsum = (float*) alloc((size_t)4 * Hn * sizeof(float));
  bf16_t* Xbf  = (bf16_t*)alloc((size_t)Tn * Bn * In * sizeof(bf16_t));
  float*  gts  = (float*) alloc((size_t)Bn * 4 * Hn * sizeof(float));
  float*  c    = (float*) alloc((size_t)Bn * Hn * sizeof(float));
  float*  h    = (float*) alloc((size_t)Bn * Hn * sizeof(float));
  bf16_t* h_bf = (bf16_t*)alloc((size_t)Bn * Hn * sizeof(bf16_t));
  bf16_t* z_bf = (bf16_t*)alloc((size_t)Bn * 2 * Hn * sizeof(bf16_t));

  // Zero-init recurrent state (graph-capture-safe)
  (void)hipMemsetAsync(c,    0, (size_t)Bn * Hn * sizeof(float),  stream);
  (void)hipMemsetAsync(h_bf, 0, (size_t)Bn * Hn * sizeof(bf16_t), stream);

  // One-time prep: weight concat/convert, input transpose/convert
  {
    const int total = 4 * Hn * KC;
    prep_wcat_kernel<<<(total + 255) / 256, 256, 0, stream>>>(
        W_ih, W_hh, b_ih, b_hh, Wcat, bsum);
    const int nfc = On * 2 * Hn;
    cvt_f32_to_bf16_kernel<<<(nfc + 255) / 256, 256, 0, stream>>>(W_fc, Wfc, nfc);
    const int nx = Tn * Bn * In;
    prep_x_kernel<<<(nx + 255) / 256, 256, 0, stream>>>(st, Xbf);
  }

  const dim3 blk(32, 8);
  const dim3 grd_gates(4 * Hn / 256, Bn / 32);  // N=4096, M=256
  const dim3 grd_fc(On / 256, Bn / 32);         // N=256,  M=256

  for (int t = 0; t < Tn; ++t) {
    // gates = [x_t | h] @ [W_ih | W_hh]^T + (b_ih + b_hh)
    wmma_gemm_kernel<<<grd_gates, blk, 0, stream>>>(
        Xbf + (size_t)t * Bn * In, In, In,
        h_bf, Hn, Hn,
        Wcat, KC, bsum, gts, 4 * Hn);

    // elementwise cell update
    lstm_cell_kernel<<<(Bn * Hn) / 256, 256, 0, stream>>>(gts, c, h, h_bf, z_bf);

    // attention: scores, softmax (-> probs out), ctx -> z[:,0:H]
    attn_kernel<<<Bn, 256, 0, stream>>>(h, enc, probs, z_bf, t);

    // out_t = [ctx | h] @ W_fc^T + b_fc  -> out[:, t, :]
    wmma_gemm_kernel<<<grd_fc, blk, 0, stream>>>(
        z_bf, 2 * Hn, 2 * Hn,
        z_bf, 2 * Hn, 0,
        Wfc, 2 * Hn, b_fc, out + (size_t)t * On, Tn * On);
  }
}
